// TorchaxBaseLinearLayerWithLoRA_42966852829757
// MI455X (gfx1250) — matrix operand order repaired and checked by reading, not compiled, not run on hardware
//
#include <hip/hip_runtime.h>
#include <hip/hip_bf16.h>

typedef __attribute__((ext_vector_type(16))) __bf16       v16bf;
typedef __attribute__((ext_vector_type(8)))  float        v8f;
typedef __attribute__((ext_vector_type(4)))  float        f4;
typedef __attribute__((ext_vector_type(4)))  unsigned int u4;
typedef __attribute__((ext_vector_type(2)))  unsigned int u2;
typedef __attribute__((ext_vector_type(4)))  unsigned int v4u;
typedef __attribute__((ext_vector_type(8)))  int          v8i;
typedef __attribute__((ext_vector_type(4)))  int          v4i;

#define T_TOK  8192
#define D_IN   4096
#define D_OUT  4096
#define RANK   16

#define BM 128
#define BN 128
#define BK 64
#define LDK 72   // padded LDS row stride in bf16 elements (144 B, 16B-aligned)

#if defined(__has_builtin)
#if __has_builtin(__builtin_amdgcn_tensor_load_to_lds) && __has_builtin(__builtin_amdgcn_s_wait_tensorcnt)
#define HAVE_TDM 1
#endif
#endif
#ifndef HAVE_TDM
#define HAVE_TDM 0
#endif

__device__ __forceinline__ unsigned short f32_to_bf16_rne(float f) {
    unsigned int u = __float_as_uint(f);
    unsigned int r = u + 0x7FFFu + ((u >> 16) & 1u);
    return (unsigned short)(r >> 16);
}
__device__ __forceinline__ unsigned int pack_bf16x2(float a, float b) {
    return (unsigned int)f32_to_bf16_rne(a) | ((unsigned int)f32_to_bf16_rne(b) << 16);
}
__device__ __forceinline__ float dot4f(f4 a, f4 b) {
    return a.x * b.x + a.y * b.y + a.z * b.z + a.w * b.w;
}

// ---------------------------------------------------------------------------
// f32 -> bf16 bulk converter (8 elements / thread)
// ---------------------------------------------------------------------------
__global__ __launch_bounds__(256) void cvt_f32_bf16_kernel(
    const float* __restrict__ in, unsigned short* __restrict__ outp, int n8)
{
    int i = blockIdx.x * blockDim.x + threadIdx.x;
    if (i >= n8) return;
    const f4* p = (const f4*)(in + (size_t)i * 8);
    f4 a = p[0], b = p[1];
    u4 o;
    o.x = pack_bf16x2(a.x, a.y);
    o.y = pack_bf16x2(a.z, a.w);
    o.z = pack_bf16x2(b.x, b.y);
    o.w = pack_bf16x2(b.z, b.w);
    *(u4*)(outp + (size_t)i * 8) = o;
}

// ---------------------------------------------------------------------------
// LoRA shrink: one wave32 per token, s[t,r] = x[t,:] . lora_a[l,r,:]
// ---------------------------------------------------------------------------
__global__ __launch_bounds__(256) void lora_shrink_kernel(
    const float* __restrict__ x,
    const float* __restrict__ lora_a,
    const int*   __restrict__ token_lora_idx,
    float*       __restrict__ s)
{
    const int lane = threadIdx.x & 31;
    const int t    = blockIdx.x * (blockDim.x >> 5) + (threadIdx.x >> 5);
    if (t >= T_TOK) return;

    const int l = token_lora_idx[t];
    const float* xr = x + (size_t)t * D_IN;
    const float* ar = lora_a + (size_t)l * RANK * D_IN;

    float part[RANK];
#pragma unroll
    for (int r = 0; r < RANK; ++r) part[r] = 0.0f;

    for (int d = lane * 4; d < D_IN; d += 32 * 4) {
        f4 xv = *(const f4*)(xr + d);
#pragma unroll
        for (int r = 0; r < RANK; ++r) {
            f4 av = *(const f4*)(ar + (size_t)r * D_IN + d);
            part[r] += dot4f(xv, av);
        }
    }
#pragma unroll
    for (int r = 0; r < RANK; ++r) {
        float v = part[r];
#pragma unroll
        for (int off = 16; off > 0; off >>= 1) v += __shfl_xor(v, off, 32);
        if (lane == 0) s[(size_t)t * RANK + r] = v;
    }
}

// ---------------------------------------------------------------------------
// One 32-deep WMMA step over a staged LDS tile (pass sA/sB pre-offset by k).
// A (16x32): lanes 0-15 -> M=lane, K {0..7,16..23}; lanes 16-31 -> K {8..15,24..31}
// B (32x16): lanes 0-15 -> N=lane, K {0..7,16..23}; lanes 16-31 -> K {8..15,24..31}
// ---------------------------------------------------------------------------
__device__ __forceinline__ void wmma_step(const unsigned short* sA,
                                          const unsigned short* sB,
                                          int waveM, int waveN, int lane,
                                          v8f acc[2][4])
{
    const int frow   = lane & 15;
    const int kheadA = (lane < 16) ? 0 : 8;
    const int kheadB = (lane < 16) ? 0 : 8;

    v16bf afr[2], bfr[4];
#pragma unroll
    for (int mt = 0; mt < 2; ++mt) {
        const int row = waveM * 32 + mt * 16 + frow;
        union { u4 q[2]; v16bf v; } f;
        f.q[0] = *(const u4*)(sA + row * LDK + kheadA);
        f.q[1] = *(const u4*)(sA + row * LDK + kheadA + 16);
        afr[mt] = f.v;
    }
#pragma unroll
    for (int nt = 0; nt < 4; ++nt) {
        const int row = waveN * 64 + nt * 16 + frow;
        union { u4 q[2]; v16bf v; } f;
        f.q[0] = *(const u4*)(sB + row * LDK + kheadB);
        f.q[1] = *(const u4*)(sB + row * LDK + kheadB + 16);
        bfr[nt] = f.v;
    }
#pragma unroll
    for (int mt = 0; mt < 2; ++mt)
#pragma unroll
        for (int nt = 0; nt < 4; ++nt)
            acc[mt][nt] = __builtin_amdgcn_wmma_f32_16x16x32_bf16(
                false, afr[mt], false, bfr[nt],
                (short)0, acc[mt][nt], false, false);
}

// Epilogue: fused LoRA expand + store.
// C/D layout: VGPR v, lanes 0-15 -> M=v, N=lane; lanes 16-31 -> M=8+v, N=lane-16
__device__ __forceinline__ void epilogue_store(v8f acc[2][4],
        const float* __restrict__ lora_b, const int* __restrict__ token_lora_idx,
        const float* __restrict__ s, float* __restrict__ out,
        int bM, int bN, int waveM, int waveN, int lane)
{
    const int N     = lane & 15;
    const int Mbase = (lane >> 4) << 3;
#pragma unroll
    for (int mt = 0; mt < 2; ++mt) {
#pragma unroll
        for (int nt = 0; nt < 4; ++nt) {
            const int o     = bN + waveN * 64 + nt * 16 + N;
            const int trow0 = bM + waveM * 32 + mt * 16 + Mbase;
#pragma unroll
            for (int v = 0; v < 8; ++v) {
                const int t = trow0 + v;
                const int l = token_lora_idx[t];
                const f4* br = (const f4*)(lora_b + ((size_t)l * D_OUT + o) * RANK);
                const f4* sr = (const f4*)(s + (size_t)t * RANK);
                float add = dot4f(sr[0], br[0]) + dot4f(sr[1], br[1]) +
                            dot4f(sr[2], br[2]) + dot4f(sr[3], br[3]);
                out[(size_t)t * D_OUT + o] = acc[mt][nt][v] + add;
            }
        }
    }
}

#if HAVE_TDM
// ---------------------------------------------------------------------------
// TDM descriptor + issue: 2D tile of 128 rows x 64 bf16 from a row-major
// [rows, 4096] bf16 tensor into LDS, padding 4 dwords every 32 dwords
// (one 128B row) => LDS row stride 72 elements.
// ---------------------------------------------------------------------------
__device__ __forceinline__ void tdm_load_tile(unsigned long long gaddr,
                                              unsigned lds_off,
                                              unsigned tensor_rows)
{
    v4u g0;
    g0.x = 1u;                                            // count=1, user mode
    g0.y = lds_off;                                       // LDS byte address
    g0.z = (unsigned)(gaddr & 0xFFFFFFFFull);             // global_addr[31:0]
    g0.w = (unsigned)((gaddr >> 32) & 0x1FFFFFFull)       // global_addr[56:32]
         | (2u << 30);                                    // type=2 (image)
    v8i g1;
    g1[0] = (1 << 16)      // data_size = 2 bytes
          | (1 << 20)      // pad_enable
          | (4 << 22)      // pad_interval: 32 dwords (one 128B row)
          | (3 << 25);     // pad_amount: 4 dwords (16B -> stride 72 elems)
    g1[1] = (D_IN & 0xFFFF) << 16;                        // tensor_dim0[15:0]
    g1[2] = (D_IN >> 16) | ((int)(tensor_rows & 0xFFFF) << 16); // dim0 hi | dim1 lo
    g1[3] = (int)(tensor_rows >> 16) | (BK << 16);        // dim1 hi | tile_dim0=64
    g1[4] = BM;                                           // tile_dim1=128, tile_dim2=0
    g1[5] = D_IN;                                         // tensor_dim0_stride lo
    g1[6] = 0;                                            // stride hi | dim1_stride lo
    g1[7] = 0;
    v4i gz = {0, 0, 0, 0};
#if __clang_major__ >= 23
    v8i gz8 = {0, 0, 0, 0, 0, 0, 0, 0};
    __builtin_amdgcn_tensor_load_to_lds(g0, g1, gz, gz, gz8, 0);
#else
    __builtin_amdgcn_tensor_load_to_lds(g0, g1, gz, gz, 0);
#endif
}

// bf16 GEMM, TDM double-buffered staging.
// smem: [A0 B0 A1 B1]; A/B tile = 128*72*2 = 18432 B, buffer pair = 36864 B.
__global__ __launch_bounds__(256) void lora_gemm_bf16_tdm(
    const unsigned short* __restrict__ xh,
    const unsigned short* __restrict__ wh,
    const float* __restrict__ lora_b,
    const int*   __restrict__ token_lora_idx,
    const float* __restrict__ s,
    float*       __restrict__ out)
{
    __shared__ unsigned short smem[2 * (BM + BN) * LDK];   // 73728 B

    const int tid   = threadIdx.x;
    const int lane  = tid & 31;
    const int wave  = tid >> 5;
    const int waveM = wave & 3;
    const int waveN = wave >> 2;
    const int bM    = blockIdx.y * BM;
    const int bN    = blockIdx.x * BN;

    const unsigned lds_base = (unsigned)(unsigned long long)(&smem[0]);
    const unsigned long long gA0 = (unsigned long long)xh + ((size_t)bM * D_IN) * 2ull;
    const unsigned long long gB0 = (unsigned long long)wh + ((size_t)bN * D_IN) * 2ull;

    v8f acc[2][4];
    const v8f vzero = {0.f, 0.f, 0.f, 0.f, 0.f, 0.f, 0.f, 0.f};
#pragma unroll
    for (int i = 0; i < 2; ++i)
#pragma unroll
        for (int j = 0; j < 4; ++j) acc[i][j] = vzero;

    // Prologue: wave 0 DMAs the first K-tile into buffer 0.
    if (wave == 0) {
        tdm_load_tile(gA0, lds_base +      0u, T_TOK);
        tdm_load_tile(gB0, lds_base + 18432u, D_OUT);
        __builtin_amdgcn_s_wait_tensorcnt(0);
    }
    __syncthreads();

    int cur = 0;
    for (int kb = 0; kb < D_IN; kb += BK) {
        const int nxt = cur ^ 1;
        if (wave == 0 && (kb + BK) < D_IN) {
            tdm_load_tile(gA0 + (size_t)(kb + BK) * 2ull,
                          lds_base + (unsigned)nxt * 36864u, T_TOK);
            tdm_load_tile(gB0 + (size_t)(kb + BK) * 2ull,
                          lds_base + (unsigned)nxt * 36864u + 18432u, D_OUT);
        }

        const unsigned short* sA = smem + cur * ((BM + BN) * LDK);
        const unsigned short* sB = sA + BM * LDK;
        wmma_step(sA,      sB,      waveM, waveN, lane, acc);   // K 0..31
        wmma_step(sA + 32, sB + 32, waveM, waveN, lane, acc);   // K 32..63

        if (wave == 0) __builtin_amdgcn_s_wait_tensorcnt(0);
        __syncthreads();
        cur = nxt;
    }

    epilogue_store(acc, lora_b, token_lora_idx, s, out, bM, bN, waveM, waveN, lane);
}
#else
// bf16 GEMM, plain vector staging (no TDM builtin available).
__global__ __launch_bounds__(256) void lora_gemm_bf16_vec(
    const unsigned short* __restrict__ xh,
    const unsigned short* __restrict__ wh,
    const float* __restrict__ lora_b,
    const int*   __restrict__ token_lora_idx,
    const float* __restrict__ s,
    float*       __restrict__ out)
{
    __shared__ unsigned short sA[BM * LDK];
    __shared__ unsigned short sB[BN * LDK];

    const int tid   = threadIdx.x;
    const int lane  = tid & 31;
    const int wave  = tid >> 5;
    const int waveM = wave & 3;
    const int waveN = wave >> 2;
    const int bM    = blockIdx.y * BM;
    const int bN    = blockIdx.x * BN;

    v8f acc[2][4];
    const v8f vzero = {0.f, 0.f, 0.f, 0.f, 0.f, 0.f, 0.f, 0.f};
#pragma unroll
    for (int i = 0; i < 2; ++i)
#pragma unroll
        for (int j = 0; j < 4; ++j) acc[i][j] = vzero;

    const int lrow = tid >> 3;        // 0..31
    const int lcol = (tid & 7) * 8;   // 0,8,...,56

    for (int kb = 0; kb < D_IN; kb += BK) {
#pragma unroll
        for (int it = 0; it < 4; ++it) {
            const int r = lrow + it * 32;
            *(u4*)(&sA[r * LDK + lcol]) =
                *(const u4*)(xh + (size_t)(bM + r) * D_IN + kb + lcol);
            *(u4*)(&sB[r * LDK + lcol]) =
                *(const u4*)(wh + (size_t)(bN + r) * D_IN + kb + lcol);
        }
        __syncthreads();
        wmma_step(sA,      sB,      waveM, waveN, lane, acc);
        wmma_step(sA + 32, sB + 32, waveM, waveN, lane, acc);
        __syncthreads();
    }

    epilogue_store(acc, lora_b, token_lora_idx, s, out, bM, bN, waveM, waveN, lane);
}
#endif

// ---------------------------------------------------------------------------
// Fallback: f32 GEMM with in-loop bf16 conversion (used if ws too small)
// ---------------------------------------------------------------------------
__global__ __launch_bounds__(256) void lora_gemm_f32_kernel(
    const float* __restrict__ x,
    const float* __restrict__ W,
    const float* __restrict__ lora_b,
    const int*   __restrict__ token_lora_idx,
    const float* __restrict__ s,
    float*       __restrict__ out)
{
    __shared__ unsigned short sA[BM * LDK];
    __shared__ unsigned short sB[BN * LDK];

    const int tid   = threadIdx.x;
    const int lane  = tid & 31;
    const int wave  = tid >> 5;
    const int waveM = wave & 3;
    const int waveN = wave >> 2;
    const int bM    = blockIdx.y * BM;
    const int bN    = blockIdx.x * BN;

    v8f acc[2][4];
    const v8f vzero = {0.f, 0.f, 0.f, 0.f, 0.f, 0.f, 0.f, 0.f};
#pragma unroll
    for (int i = 0; i < 2; ++i)
#pragma unroll
        for (int j = 0; j < 4; ++j) acc[i][j] = vzero;

    const int lrow = tid >> 4;         // 0..15
    const int lcol = (tid & 15) * 4;   // 0..60

    for (int kb = 0; kb < D_IN; kb += BK) {
#pragma unroll
        for (int it = 0; it < 8; ++it) {
            const int r = lrow + it * 16;
            f4 av = *(const f4*)(x + (size_t)(bM + r) * D_IN + kb + lcol);
            f4 bv = *(const f4*)(W + (size_t)(bN + r) * D_IN + kb + lcol);
            u2 ap, bp;
            ap.x = pack_bf16x2(av.x, av.y);
            ap.y = pack_bf16x2(av.z, av.w);
            bp.x = pack_bf16x2(bv.x, bv.y);
            bp.y = pack_bf16x2(bv.z, bv.w);
            *(u2*)(&sA[r * LDK + lcol]) = ap;
            *(u2*)(&sB[r * LDK + lcol]) = bp;
        }
        __syncthreads();
        wmma_step(sA,      sB,      waveM, waveN, lane, acc);
        wmma_step(sA + 32, sB + 32, waveM, waveN, lane, acc);
        __syncthreads();
    }

    epilogue_store(acc, lora_b, token_lora_idx, s, out, bM, bN, waveM, waveN, lane);
}

// ---------------------------------------------------------------------------
extern "C" void kernel_launch(void* const* d_in, const int* in_sizes, int n_in,
                              void* d_out, int out_size, void* d_ws, size_t ws_size,
                              hipStream_t stream) {
    const float* x       = (const float*)d_in[0];
    const float* W       = (const float*)d_in[1];
    const float* lora_a  = (const float*)d_in[2];
    const float* lora_b  = (const float*)d_in[3];
    const int*   tidx    = (const int*)d_in[4];
    float*       out     = (float*)d_out;

    const size_t S_BYTES  = (size_t)T_TOK * RANK * sizeof(float);          // 512 KB
    const size_t XH_ELEMS = (size_t)T_TOK * D_IN;
    const size_t WH_ELEMS = (size_t)D_OUT * D_IN;
    const size_t NEED     = S_BYTES + (XH_ELEMS + WH_ELEMS) * sizeof(unsigned short);

    float* s = (float*)d_ws;
    lora_shrink_kernel<<<dim3(T_TOK / 8), 256, 0, stream>>>(x, lora_a, tidx, s);

    dim3 grid(D_OUT / BN, T_TOK / BM);   // (32, 64)

    if (ws_size >= NEED) {
        unsigned short* xh = (unsigned short*)((char*)d_ws + S_BYTES);
        unsigned short* wh = xh + XH_ELEMS;
        cvt_f32_bf16_kernel<<<dim3((unsigned)(XH_ELEMS / 8 / 256)), 256, 0, stream>>>(
            x, xh, (int)(XH_ELEMS / 8));
        cvt_f32_bf16_kernel<<<dim3((unsigned)(WH_ELEMS / 8 / 256)), 256, 0, stream>>>(
            W, wh, (int)(WH_ELEMS / 8));
#if HAVE_TDM
        lora_gemm_bf16_tdm<<<grid, 256, 0, stream>>>(xh, wh, lora_b, tidx, s, out);
#else
        lora_gemm_bf16_vec<<<grid, 256, 0, stream>>>(xh, wh, lora_b, tidx, s, out);
#endif
    } else {
        lora_gemm_f32_kernel<<<grid, 256, 0, stream>>>(x, W, lora_b, tidx, s, out);
    }
}
